// SpatialAxialAttention_463856468703
// MI455X (gfx1250) — matrix operand order repaired and checked by reading, not compiled, not run on hardware
//
#include <hip/hip_runtime.h>
#include <hip/hip_bf16.h>

typedef __attribute__((ext_vector_type(16))) __bf16 v16bf;
typedef __attribute__((ext_vector_type(8)))  float  v8f;
typedef __attribute__((ext_vector_type(4)))  int    v4i;

#define HEADS  16
#define HD     64
#define DMODEL 1024
#define NQKV   3072
#define ROWS   9216   // B*T*H*W
#define SEQ    576    // H*W
#define FRAMES 16
#define HH_    18
#define PI_F   3.14159265358979323846f

// Async global->LDS DMA (CDNA5): gate on builtin availability, fall back to
// the register-staging path otherwise.
#if defined(__has_builtin)
#  if __has_builtin(__builtin_amdgcn_global_load_async_to_lds_b128) && \
      __has_builtin(__builtin_amdgcn_s_wait_asynccnt)
#    define USE_ASYNC_LDS 1
#  endif
#endif
#ifndef USE_ASYNC_LDS
#  define USE_ASYNC_LDS 0
#endif

union V16 { v16bf v; __bf16 e[16]; uint4 q[2]; };
union V8  { v8f   v; float  e[8];  };
union BF4 { uint2 u; __bf16 e[4]; };
union BF8 { uint4 u; __bf16 e[8]; };

__device__ __forceinline__ v8f wmma_bf16(v16bf a, v16bf b, v8f c) {
  // D = A(16x32 bf16) x B(32x16 bf16) + C(16x16 f32)
  return __builtin_amdgcn_wmma_f32_16x16x32_bf16(false, a, false, b, (short)0, c,
                                                 false, false);
}

__device__ __forceinline__ uint2 pack4bf(float4 f) {
  BF4 p;
  p.e[0] = (__bf16)f.x; p.e[1] = (__bf16)f.y;
  p.e[2] = (__bf16)f.z; p.e[3] = (__bf16)f.w;
  return p.u;
}

#if USE_ASYNC_LDS
__device__ __forceinline__ void async_copy_b128(const void* g, void* l) {
  __builtin_amdgcn_global_load_async_to_lds_b128(
      (__attribute__((address_space(1))) v4i*)g,
      (__attribute__((address_space(3))) v4i*)l, 0, 0);
}
#endif

// ---------------------------------------------------------------------------
// Kernel 1: qkv = x @ Wqkv  (bf16 WMMA, f32 accum), fused axial RoPE on q/k,
// scatter into Q/K/V [frame][head][n][d] bf16 for the attention kernel.
// 64x64 tile / block, 4 waves, K pipelined in chunks of 32 (register double
// buffer; both tiles need f32->bf16 conversion so they stay on the VALU path).
// ---------------------------------------------------------------------------
__global__ __launch_bounds__(128)
void qkv_rope_kernel(const float* __restrict__ X, const float* __restrict__ Wq,
                     __bf16* __restrict__ Qb, __bf16* __restrict__ Kb,
                     __bf16* __restrict__ Vb)
{
  __shared__ __bf16 xs[64][32];   // A tile: rows x k
  __shared__ __bf16 wt[64][32];   // B tile transposed: n x k
  const int rowBase = blockIdx.x * 64;
  const int colBase = blockIdx.y * 64;
  const int tid  = threadIdx.x;
  const int wave = tid >> 5, lane = tid & 31;
  const int lrow = lane & 15, lhi = lane >> 4;

  V8 acc[4] = {};
  float4 xr[4], wr[4];

#pragma unroll
  for (int i = 0; i < 4; ++i) {
    int idx = tid + i * 128;                       // 512 float4 per tile
    xr[i] = *(const float4*)&X[(size_t)(rowBase + (idx >> 3)) * DMODEL + ((idx & 7) << 2)];
    wr[i] = *(const float4*)&Wq[(size_t)(idx >> 4) * NQKV + colBase + ((idx & 15) << 2)];
  }

  for (int kt = 0; kt < DMODEL; kt += 32) {
    __syncthreads();
#pragma unroll
    for (int i = 0; i < 4; ++i) {
      int idx = tid + i * 128;
      int m = idx >> 3, k = (idx & 7) << 2;
      *(uint2*)&xs[m][k] = pack4bf(xr[i]);
    }
#pragma unroll
    for (int i = 0; i < 4; ++i) {
      int idx = tid + i * 128;
      int k = idx >> 4, n = (idx & 15) << 2;
      wt[n + 0][k] = (__bf16)wr[i].x;
      wt[n + 1][k] = (__bf16)wr[i].y;
      wt[n + 2][k] = (__bf16)wr[i].z;
      wt[n + 3][k] = (__bf16)wr[i].w;
    }
    __syncthreads();

    if (kt + 32 < DMODEL) {
#pragma unroll
      for (int i = 0; i < 4; ++i) {
        int idx = tid + i * 128;
        xr[i] = *(const float4*)&X[(size_t)(rowBase + (idx >> 3)) * DMODEL + kt + 32 + ((idx & 7) << 2)];
        wr[i] = *(const float4*)&Wq[(size_t)(kt + 32 + (idx >> 4)) * NQKV + colBase + ((idx & 15) << 2)];
      }
    }

    V16 a;
    a.q[0] = *(const uint4*)&xs[wave * 16 + lrow][lhi * 8];
    a.q[1] = *(const uint4*)&xs[wave * 16 + lrow][16 + lhi * 8];
    V16 b[4];
#pragma unroll
    for (int s = 0; s < 4; ++s) {
      b[s].q[0] = *(const uint4*)&wt[s * 16 + lrow][lhi * 8];
      b[s].q[1] = *(const uint4*)&wt[s * 16 + lrow][16 + lhi * 8];
    }
#pragma unroll
    for (int s = 0; s < 4; ++s) acc[s].v = wmma_bf16(a.v, b[s].v, acc[s].v);
  }

  // Epilogue: section 0=q, 1=k, 2=v (64-wide tile never straddles sections)
  const int sec    = colBase >> 10;
  const int ccBase = colBase & 1023;
  __bf16* dst = (sec == 0) ? Qb : (sec == 1) ? Kb : Vb;

#pragma unroll
  for (int s = 0; s < 4; ++s) {
    int cc   = ccBase + s * 16 + lrow;
    int head = cc >> 6, d = cc & 63;
#pragma unroll
    for (int j = 0; j < 8; ++j) {
      int r = rowBase + wave * 16 + j + lhi * 8;
      float val = acc[s].e[j];
      if (sec < 2) {
        // axial RoPE: d<32 uses H position, d>=32 uses W position
        int hh = (r >> 5) % HH_;
        int ww = r & 31;
        float f;
        if (d < 32) {
          float band = (1.0f + 127.0f * (float)(d >> 1) * (1.0f / 15.0f)) * PI_F;
          float pos  = -1.0f + 2.0f * (float)hh * (1.0f / 17.0f);
          f = pos * band;
        } else {
          float band = (1.0f + 127.0f * (float)((d - 32) >> 1) * (1.0f / 15.0f)) * PI_F;
          float pos  = -1.0f + 2.0f * (float)ww * (1.0f / 31.0f);
          f = pos * band;
        }
        float cs = __cosf(f), sn = __sinf(f);
        float other = __shfl_xor(val, 1, 32);   // pair partner = adjacent column
        val = (d & 1) ? (val * cs + other * sn) : (val * cs - other * sn);
      }
      int fr = r / SEQ;
      int n  = r - fr * SEQ;
      dst[(((size_t)fr * HEADS + head) * SEQ + n) * HD + d] = (__bf16)val;
    }
  }
}

// ---------------------------------------------------------------------------
// Kernel 2: flash attention per (frame, head, 64-query-row tile).
// 4 waves/block, each wave owns 16 query rows. K tile: async global->LDS DMA
// into a double buffer (ASYNCcnt). V tile: register path (needs transpose).
// ---------------------------------------------------------------------------
__global__ __launch_bounds__(128)
void attention_kernel(const __bf16* __restrict__ Qb, const __bf16* __restrict__ Kb,
                      const __bf16* __restrict__ Vb, __bf16* __restrict__ att)
{
  __shared__ __bf16 sK[2][64][64];     // [buf][key_rel][d]
  __shared__ __bf16 sVt[64][64];       // [d][key_rel]
  __shared__ __bf16 sP[4][16][64];     // per-wave P staging (C->A relayout)

  const int qt   = blockIdx.x;   // 0..8
  const int head = blockIdx.y;   // 0..15
  const int fr   = blockIdx.z;   // 0..15
  const int tid  = threadIdx.x;
  const int wave = tid >> 5, lane = tid & 31;
  const int lrow = lane & 15, lhi = lane >> 4;

  const size_t hb = ((size_t)fr * HEADS + head) * SEQ * HD;
  const __bf16* Qh = Qb + hb;
  const __bf16* Kh = Kb + hb;
  const __bf16* Vh = Vb + hb;

  // Q fragments: 16 rows per wave, K=64 as two 16x32 A-fragments
  V16 qa[2];
  {
    int qrow = qt * 64 + wave * 16 + lrow;
#pragma unroll
    for (int ks = 0; ks < 2; ++ks) {
      qa[ks].q[0] = *(const uint4*)&Qh[(size_t)qrow * HD + ks * 32 + lhi * 8];
      qa[ks].q[1] = *(const uint4*)&Qh[(size_t)qrow * HD + ks * 32 + 16 + lhi * 8];
    }
  }

  V8 O[4] = {};
  float mrow[8], lsum[8];
#pragma unroll
  for (int j = 0; j < 8; ++j) { mrow[j] = -1e30f; lsum[j] = 0.0f; }

#if USE_ASYNC_LDS
  // issue async DMA of K tile t into LDS buffer buf (4 x b128 per lane)
  auto asyncK = [&](int t, int buf) {
#pragma unroll
    for (int i = 0; i < 4; ++i) {
      int idx = tid + i * 128;
      int kr = idx >> 3, d8 = (idx & 7) << 3;
      async_copy_b128(&Kh[(size_t)(t * 64 + kr) * HD + d8], &sK[buf][kr][d8]);
    }
  };
  asyncK(0, 0);
#else
  uint4 kreg[4];
#pragma unroll
  for (int i = 0; i < 4; ++i) {
    int idx = tid + i * 128;
    kreg[i] = *(const uint4*)&Kh[(size_t)(idx >> 3) * HD + ((idx & 7) << 3)];
  }
#endif
  BF8 vreg[4];
#pragma unroll
  for (int i = 0; i < 4; ++i) {
    int idx = tid + i * 128;
    vreg[i].u = *(const uint4*)&Vh[(size_t)(idx >> 3) * HD + ((idx & 7) << 3)];
  }

  for (int t = 0; t < SEQ / 64; ++t) {
    const int cur = t & 1;
    __syncthreads();                       // prior tile fully consumed
#pragma unroll
    for (int i = 0; i < 4; ++i) {
      int idx = tid + i * 128;
      int kr = idx >> 3, d8 = (idx & 7) << 3;
#if !USE_ASYNC_LDS
      *(uint4*)&sK[cur][kr][d8] = kreg[i];
#endif
#pragma unroll
      for (int j = 0; j < 8; ++j) sVt[d8 + j][kr] = vreg[i].e[j];   // transpose V
    }
#if USE_ASYNC_LDS
    __builtin_amdgcn_s_wait_asynccnt(0);   // K tile t landed in sK[cur]
#endif
    __syncthreads();                       // all staging visible to all waves

    if (t + 1 < SEQ / 64) {
#if USE_ASYNC_LDS
      asyncK(t + 1, cur ^ 1);              // DMA next K tile during compute
#else
#pragma unroll
      for (int i = 0; i < 4; ++i) {
        int idx = tid + i * 128;
        kreg[i] = *(const uint4*)&Kh[(size_t)((t + 1) * 64 + (idx >> 3)) * HD + ((idx & 7) << 3)];
      }
#endif
#pragma unroll
      for (int i = 0; i < 4; ++i) {
        int idx = tid + i * 128;
        vreg[i].u = *(const uint4*)&Vh[(size_t)((t + 1) * 64 + (idx >> 3)) * HD + ((idx & 7) << 3)];
      }
    }

    // S = Q @ K^T  (16 x 64 per wave)
    V8 S[4] = {};
    V16 bk[4];
#pragma unroll
    for (int ks = 0; ks < 2; ++ks) {
#pragma unroll
      for (int ns = 0; ns < 4; ++ns) {
        bk[ns].q[0] = *(const uint4*)&sK[cur][ns * 16 + lrow][ks * 32 + lhi * 8];
        bk[ns].q[1] = *(const uint4*)&sK[cur][ns * 16 + lrow][ks * 32 + 16 + lhi * 8];
      }
#pragma unroll
      for (int ns = 0; ns < 4; ++ns)
        S[ns].v = wmma_bf16(qa[ks].v, bk[ns].v, S[ns].v);
    }
#pragma unroll
    for (int ns = 0; ns < 4; ++ns)
#pragma unroll
      for (int j = 0; j < 8; ++j) S[ns].e[j] *= 0.125f;   // 1/sqrt(64)

    // online softmax per row (rows j / j+8 live in the two lane halves)
#pragma unroll
    for (int j = 0; j < 8; ++j) {
      float mx = S[0].e[j];
#pragma unroll
      for (int ns = 1; ns < 4; ++ns) mx = fmaxf(mx, S[ns].e[j]);
      mx = fmaxf(mx, __shfl_xor(mx, 1, 32));
      mx = fmaxf(mx, __shfl_xor(mx, 2, 32));
      mx = fmaxf(mx, __shfl_xor(mx, 4, 32));
      mx = fmaxf(mx, __shfl_xor(mx, 8, 32));
      float newm = fmaxf(mrow[j], mx);
      float corr = __expf(mrow[j] - newm);
      float sum = 0.0f;
#pragma unroll
      for (int ns = 0; ns < 4; ++ns) {
        S[ns].e[j] = __expf(S[ns].e[j] - newm);
        sum += S[ns].e[j];
      }
      sum += __shfl_xor(sum, 1, 32);
      sum += __shfl_xor(sum, 2, 32);
      sum += __shfl_xor(sum, 4, 32);
      sum += __shfl_xor(sum, 8, 32);
      lsum[j] = lsum[j] * corr + sum;
#pragma unroll
      for (int ds = 0; ds < 4; ++ds) O[ds].e[j] *= corr;
      mrow[j] = newm;
    }

    // P: C-layout -> LDS -> A-layout
#pragma unroll
    for (int ns = 0; ns < 4; ++ns)
#pragma unroll
      for (int j = 0; j < 8; ++j)
        sP[wave][j + lhi * 8][ns * 16 + lrow] = (__bf16)S[ns].e[j];
    __syncthreads();

    // O += P @ V
#pragma unroll
    for (int kc = 0; kc < 2; ++kc) {
      V16 a;
      a.q[0] = *(const uint4*)&sP[wave][lrow][kc * 32 + lhi * 8];
      a.q[1] = *(const uint4*)&sP[wave][lrow][kc * 32 + 16 + lhi * 8];
      V16 bv[4];
#pragma unroll
      for (int ds = 0; ds < 4; ++ds) {
        bv[ds].q[0] = *(const uint4*)&sVt[ds * 16 + lrow][kc * 32 + lhi * 8];
        bv[ds].q[1] = *(const uint4*)&sVt[ds * 16 + lrow][kc * 32 + 16 + lhi * 8];
      }
#pragma unroll
      for (int ds = 0; ds < 4; ++ds)
        O[ds].v = wmma_bf16(a.v, bv[ds].v, O[ds].v);
    }
  }

  // normalize and store to [frame*576 + n][head*64 + d] (bf16)
#pragma unroll
  for (int ds = 0; ds < 4; ++ds) {
    int d = ds * 16 + lrow;
#pragma unroll
    for (int j = 0; j < 8; ++j) {
      int n = qt * 64 + wave * 16 + j + lhi * 8;
      float v = O[ds].e[j] / lsum[j];
      att[((size_t)fr * SEQ + n) * (HEADS * HD) + head * HD + d] = (__bf16)v;
    }
  }
}

// ---------------------------------------------------------------------------
// Kernel 3: out = att(9216x1024 bf16) @ Wout(1024x1024) + bout  (f32 out)
// A tile: async global->LDS DMA double buffer; W tile: register path (cvt).
// ---------------------------------------------------------------------------
__global__ __launch_bounds__(128)
void out_proj_kernel(const __bf16* __restrict__ A, const float* __restrict__ Wo,
                     const float* __restrict__ bout, float* __restrict__ Y)
{
  __shared__ __bf16 xs[2][64][32];
  __shared__ __bf16 wt[64][32];
  const int rowBase = blockIdx.x * 64;
  const int colBase = blockIdx.y * 64;
  const int tid  = threadIdx.x;
  const int wave = tid >> 5, lane = tid & 31;
  const int lrow = lane & 15, lhi = lane >> 4;

  V8 acc[4] = {};
  float4 wr[4];

#if USE_ASYNC_LDS
  auto asyncA = [&](int kt, int buf) {
#pragma unroll
    for (int i = 0; i < 2; ++i) {            // 256 uint4 per A tile
      int idx = tid + i * 128;
      async_copy_b128(&A[(size_t)(rowBase + (idx >> 2)) * DMODEL + kt + ((idx & 3) << 3)],
                      &xs[buf][idx >> 2][(idx & 3) << 3]);
    }
  };
  asyncA(0, 0);
#else
  uint4 ar[2];
#pragma unroll
  for (int i = 0; i < 2; ++i) {
    int idx = tid + i * 128;
    ar[i] = *(const uint4*)&A[(size_t)(rowBase + (idx >> 2)) * DMODEL + ((idx & 3) << 3)];
  }
#endif
#pragma unroll
  for (int i = 0; i < 4; ++i) {
    int idx = tid + i * 128;
    wr[i] = *(const float4*)&Wo[(size_t)(idx >> 4) * DMODEL + colBase + ((idx & 15) << 2)];
  }

  for (int kt = 0; kt < DMODEL; kt += 32) {
    const int cur = (kt >> 5) & 1;
    __syncthreads();
#if !USE_ASYNC_LDS
#pragma unroll
    for (int i = 0; i < 2; ++i) {
      int idx = tid + i * 128;
      *(uint4*)&xs[cur][idx >> 2][(idx & 3) << 3] = ar[i];
    }
#endif
#pragma unroll
    for (int i = 0; i < 4; ++i) {
      int idx = tid + i * 128;
      int k = idx >> 4, n = (idx & 15) << 2;
      wt[n + 0][k] = (__bf16)wr[i].x;
      wt[n + 1][k] = (__bf16)wr[i].y;
      wt[n + 2][k] = (__bf16)wr[i].z;
      wt[n + 3][k] = (__bf16)wr[i].w;
    }
#if USE_ASYNC_LDS
    __builtin_amdgcn_s_wait_asynccnt(0);     // A tile landed in xs[cur]
#endif
    __syncthreads();

    if (kt + 32 < DMODEL) {
#if USE_ASYNC_LDS
      asyncA(kt + 32, cur ^ 1);
#else
#pragma unroll
      for (int i = 0; i < 2; ++i) {
        int idx = tid + i * 128;
        ar[i] = *(const uint4*)&A[(size_t)(rowBase + (idx >> 2)) * DMODEL + kt + 32 + ((idx & 3) << 3)];
      }
#endif
#pragma unroll
      for (int i = 0; i < 4; ++i) {
        int idx = tid + i * 128;
        wr[i] = *(const float4*)&Wo[(size_t)(kt + 32 + (idx >> 4)) * DMODEL + colBase + ((idx & 15) << 2)];
      }
    }

    V16 a;
    a.q[0] = *(const uint4*)&xs[cur][wave * 16 + lrow][lhi * 8];
    a.q[1] = *(const uint4*)&xs[cur][wave * 16 + lrow][16 + lhi * 8];
    V16 b[4];
#pragma unroll
    for (int s = 0; s < 4; ++s) {
      b[s].q[0] = *(const uint4*)&wt[s * 16 + lrow][lhi * 8];
      b[s].q[1] = *(const uint4*)&wt[s * 16 + lrow][16 + lhi * 8];
    }
#pragma unroll
    for (int s = 0; s < 4; ++s) acc[s].v = wmma_bf16(a.v, b[s].v, acc[s].v);
  }

#pragma unroll
  for (int s = 0; s < 4; ++s) {
    int c = colBase + s * 16 + lrow;
    float bb = bout[c];
#pragma unroll
    for (int j = 0; j < 8; ++j) {
      int r = rowBase + wave * 16 + j + lhi * 8;
      Y[(size_t)r * DMODEL + c] = acc[s].e[j] + bb;
    }
  }
}

// ---------------------------------------------------------------------------
extern "C" void kernel_launch(void* const* d_in, const int* in_sizes, int n_in,
                              void* d_out, int out_size, void* d_ws, size_t ws_size,
                              hipStream_t stream) {
  const float* x    = (const float*)d_in[0];
  const float* Wqkv = (const float*)d_in[1];
  const float* Wout = (const float*)d_in[2];
  const float* bout = (const float*)d_in[3];
  float* out = (float*)d_out;

  const size_t perTensor = (size_t)FRAMES * HEADS * SEQ * HD;  // 9,437,184 elems
  __bf16* Qb  = (__bf16*)d_ws;
  __bf16* Kb  = Qb + perTensor;
  __bf16* Vb  = Kb + perTensor;
  __bf16* att = Vb + perTensor;                                // 9216 x 1024

  qkv_rope_kernel<<<dim3(ROWS / 64, NQKV / 64), 128, 0, stream>>>(x, Wqkv, Qb, Kb, Vb);
  attention_kernel<<<dim3(SEQ / 64, HEADS, FRAMES), 128, 0, stream>>>(Qb, Kb, Vb, att);
  out_proj_kernel<<<dim3(ROWS / 64, DMODEL / 64), 128, 0, stream>>>(att, Wout, bout, out);
}
